// MOTMPNet_4329327035195
// MI455X (gfx1250) — compile-verified, matrix operand order checked by reading, not run on hardware
//
#include <hip/hip_runtime.h>

// ---------------- types ----------------
typedef __bf16 bf16_t;
typedef __attribute__((ext_vector_type(16))) __bf16 v16bf;
typedef __attribute__((ext_vector_type(8)))  __bf16 v8bf;
typedef __attribute__((ext_vector_type(8)))  float  v8f;

// ---------------- scalar helpers ----------------
static __device__ __forceinline__ bf16_t f2bf(float f) {
  unsigned u = __float_as_uint(f);
  unsigned r = (u + 0x7fffu + ((u >> 16) & 1u)) >> 16;   // RNE
  return __builtin_bit_cast(bf16_t, (unsigned short)r);
}
static __device__ __forceinline__ float bf2f(bf16_t b) {
  unsigned u = ((unsigned)__builtin_bit_cast(unsigned short, b)) << 16;
  return __uint_as_float(u);
}
// monotone encode of f32 into u32 for atomicMax-based segment max
static __device__ __forceinline__ unsigned fenc(float f) {
  unsigned u = __float_as_uint(f);
  return (u & 0x80000000u) ? ~u : (u | 0x80000000u);
}
static __device__ __forceinline__ float fdec(unsigned k) {
  unsigned u = (k & 0x80000000u) ? (k & 0x7fffffffu) : ~k;
  return __uint_as_float(u);
}

// ---------------- WMMA helpers ----------------
static __device__ __forceinline__ v8f wmma_bf16(v16bf a, v16bf b, v8f c) {
  // D(16x16 f32) = A(16x32 bf16) * B(32x16 bf16) + C
  return __builtin_amdgcn_wmma_f32_16x16x32_bf16(false, a, false, b, (short)0, c, false, false);
}

// Build a 16x32 A-fragment from a row-major bf16 LDS panel.
// rowp points at &panel[row M = lane%16][chunk K base]; h = lane/16.
// CDNA5 16-bit A layout: lanes 0-15 -> K {0..7,16..23}; lanes 16-31 -> K {8..15,24..31}
static __device__ __forceinline__ v16bf fragA(const bf16_t* rowp, int h) {
  v8bf lo = *(const v8bf*)(rowp + 8 * h);
  v8bf hi = *(const v8bf*)(rowp + 16 + 8 * h);
  v16bf r;
#pragma unroll
  for (int i = 0; i < 8; i++) { r[i] = lo[i]; r[8 + i] = hi[i]; }
  return r;
}

// Load a pre-swizzled 32x16 B-fragment (32 lanes x 16 bf16, packed) from LDS.
static __device__ __forceinline__ v16bf ldW(const bf16_t* sw, int f, int lane) {
  return *(const v16bf*)(sw + f * 512 + lane * 16);
}

// Store 8 f32 (two float4) as 8 consecutive bf16 (one 16B LDS store).
static __device__ __forceinline__ void st8(bf16_t* p, float4 a, float4 b) {
  v8bf v;
  v[0] = f2bf(a.x); v[1] = f2bf(a.y); v[2] = f2bf(a.z); v[3] = f2bf(a.w);
  v[4] = f2bf(b.x); v[5] = f2bf(b.y); v[6] = f2bf(b.z); v[7] = f2bf(b.w);
  *(v8bf*)p = v;
}
static __device__ __forceinline__ void st8z(bf16_t* p) {
  v8bf v;
#pragma unroll
  for (int i = 0; i < 8; i++) v[i] = f2bf(0.0f);
  *(v8bf*)p = v;
}

// Convert a row-major f32 weight W[K][N] into pre-swizzled bf16 B-fragments in LDS.
// Fragment f = kc*nts + nt covers K in [32*kc,32*kc+32), N in [16*nt,16*nt+16); OOB -> 0.
static __device__ void packW(bf16_t* dst, const float* __restrict__ W,
                             int K, int N, int kcs, int nts) {
  int total = kcs * nts * 512;
  for (int t = threadIdx.x; t < total; t += blockDim.x) {
    int f = t >> 9, wi = t & 511, ln = wi >> 4, i = wi & 15;
    int kc = f / nts, nt = f - kc * nts;
    int hh = ln >> 4;
    int n  = nt * 16 + (ln & 15);
    int kl = (i < 8) ? (8 * hh + i) : (8 + 8 * hh + i);  // i>=8: 16+8h+(i-8)
    int k  = kc * 32 + kl;
    dst[t] = f2bf((k < K && n < N) ? W[(size_t)k * N + n] : 0.0f);
  }
}

// ---------------- K0: zero workspace ----------------
__global__ void zero_kernel(float* p, size_t n) {
  size_t i = (size_t)blockIdx.x * blockDim.x + threadIdx.x;
  size_t s = (size_t)gridDim.x * blockDim.x;
  for (; i < n; i += s) p[i] = 0.0f;
}

// ---------------- K1: edge pipeline (WMMA-heavy) ----------------
// One wave per 16-edge tile, 4 waves per block, grid-stride over tiles.
// Weight B-fragments in LDS, indices:
//   We1: 0..14 (kc*5+nt)   We2: 15..17 (kc)
//   Wfo1: 18..25 (kc*4+nt) Wfo2: 26..29 (kc*2+nt)
//   Wfi1: 30..37 (kc*4+nt) Wfi2: 38..41 (kc*2+nt)
__global__ void __launch_bounds__(128)
edge_kernel(const float* __restrict__ x, const float* __restrict__ eattr,
            const int* __restrict__ row, const int* __restrict__ col,
            const float* __restrict__ We1, const float* __restrict__ be1,
            const float* __restrict__ We2, const float* __restrict__ be2,
            const float* __restrict__ Wfo1, const float* __restrict__ bfo1,
            const float* __restrict__ Wfo2, const float* __restrict__ bfo2,
            const float* __restrict__ Wfi1, const float* __restrict__ bfi1,
            const float* __restrict__ Wfi2, const float* __restrict__ bfi2,
            const float* __restrict__ Wc1, const float* __restrict__ bc1,
            const float* __restrict__ Wc2, const float* __restrict__ bc2,
            float* __restrict__ outE, float* __restrict__ outDec,
            float* __restrict__ flowOut, float* __restrict__ flowIn,
            unsigned* __restrict__ maxOut, unsigned* __restrict__ maxIn,
            int E) {
  __shared__ __align__(64) bf16_t sW[42 * 512];     // 43008 B, pre-swizzled B-fragments
  __shared__ __align__(64) bf16_t sH[4][16 * 96];   // per-wave activation panel (A0/h1/flow-hidden)
  __shared__ __align__(64) bf16_t sE[4][16 * 16];   // per-wave e tile (bf16)
  __shared__ int sR[4][16], sC[4][16];

  packW(sW +  0 * 512, We1, 80, 80, 3, 5);
  packW(sW + 15 * 512, We2, 80, 16, 3, 1);
  packW(sW + 18 * 512, Wfo1, 48, 64, 2, 4);
  packW(sW + 26 * 512, Wfo2, 64, 32, 2, 2);
  packW(sW + 30 * 512, Wfi1, 48, 64, 2, 4);
  packW(sW + 38 * 512, Wfi2, 64, 32, 2, 2);
  __syncthreads();

  const int tid = threadIdx.x, w = tid >> 5, lane = tid & 31, h = lane >> 4, m = lane & 15;
  const int T = (E + 15) / 16;

  for (int tb = blockIdx.x * 4; tb < T; tb += gridDim.x * 4) {
    const int tile = tb + w;
    const bool valid = tile < T;
    const int e0 = tile * 16;
    v16bf aflow0, af1;

    // ---- stage A0 = [x_row | x_col | edge_attr | 0pad] : 16x96 bf16, row-major ----
    if (valid) {
      int edge = e0 + m; if (edge >= E) edge = E - 1;
      int r = row[edge], c = col[edge];
      if (h == 0) { sR[w][m] = r; sC[w][m] = c; }
      const float4* xr = (const float4*)(x + (size_t)r * 32 + 16 * h);
      const float4* xc = (const float4*)(x + (size_t)c * 32 + 16 * h);
      const float4* ep = (const float4*)(eattr + (size_t)edge * 16 + 8 * h);
      bf16_t* ar = &sH[w][m * 96];
      st8(ar + 16 * h,          xr[0], xr[1]);
      st8(ar + 16 * h + 8,      xr[2], xr[3]);
      st8(ar + 32 + 16 * h,     xc[0], xc[1]);
      st8(ar + 32 + 16 * h + 8, xc[2], xc[3]);
      st8(ar + 64 + 8 * h,      ep[0], ep[1]);
      st8z(ar + 80 + 8 * h);
    }
    __syncthreads();

    // ---- edge MLP layer 1: 16x96(bf16) @ 96x80 -> relu -> h1 (reuse sH) ----
    if (valid) {
      const bf16_t* ar = &sH[w][m * 96];
      v16bf a0 = fragA(ar, h), a1 = fragA(ar + 32, h), a2 = fragA(ar + 64, h);
      aflow0 = a1;  // x_col chunk doubles as flow-MLP K-chunk 0; keep in registers
#pragma unroll
      for (int nt = 0; nt < 5; ++nt) {
        v8f acc = {};
        acc = wmma_bf16(a0, ldW(sW, 0 * 5 + nt, lane), acc);
        acc = wmma_bf16(a1, ldW(sW, 1 * 5 + nt, lane), acc);
        acc = wmma_bf16(a2, ldW(sW, 2 * 5 + nt, lane), acc);
        float b = be1[nt * 16 + m];
#pragma unroll
        for (int r = 0; r < 8; ++r)
          sH[w][(r + 8 * h) * 96 + nt * 16 + m] = f2bf(fmaxf(acc[r] + b, 0.0f));
      }
    }
    __syncthreads();

    // ---- edge MLP layer 2: h1 @ 80x16 -> e ----
    if (valid) {
      const bf16_t* hr = &sH[w][m * 96];
      v16bf h0 = fragA(hr, h), h1f = fragA(hr + 32, h), h2 = fragA(hr + 64, h);
      v8f acc = {};
      acc = wmma_bf16(h0,  ldW(sW, 15, lane), acc);
      acc = wmma_bf16(h1f, ldW(sW, 16, lane), acc);
      acc = wmma_bf16(h2,  ldW(sW, 17, lane), acc);
      float b = be2[m];
#pragma unroll
      for (int r = 0; r < 8; ++r) {
        float v = acc[r] + b;
        int gi = e0 + r + 8 * h;
        if (gi < E) outE[(size_t)gi * 16 + m] = v;
        sE[w][(r + 8 * h) * 16 + m] = f2bf(v);
      }
    }
    __syncthreads();

    // ---- dec classifier (16->8->1) + atomic seg-max ----
    if (valid && lane < 16 && (e0 + lane) < E) {
      float ef[16];
#pragma unroll
      for (int k = 0; k < 16; ++k) ef[k] = bf2f(sE[w][lane * 16 + k]);
      float d = bc2[0];
#pragma unroll
      for (int j = 0; j < 8; ++j) {
        float hv = bc1[j];
#pragma unroll
        for (int k = 0; k < 16; ++k) hv += ef[k] * Wc1[k * 8 + j];
        d += fmaxf(hv, 0.0f) * Wc2[j];
      }
      outDec[e0 + lane] = d;
      int r = sR[w][lane], c = sC[w][lane];
      if (r != c) {
        unsigned enc = fenc(d);
        if (r < c) atomicMax(&maxOut[r], enc); else atomicMax(&maxIn[r], enc);
      }
    }

    // ---- build flow K-chunk1 fragment from e (K 32..47 valid, 48..63 zero) ----
    if (valid) {
      v8bf lo = *(const v8bf*)(&sE[w][m * 16 + 8 * h]);
#pragma unroll
      for (int i = 0; i < 8; ++i) { af1[i] = lo[i]; af1[8 + i] = f2bf(0.0f); }
      // fo layer1: 16x64 hidden -> sH
#pragma unroll
      for (int nt = 0; nt < 4; ++nt) {
        v8f acc = {};
        acc = wmma_bf16(aflow0, ldW(sW, 18 + nt, lane), acc);
        acc = wmma_bf16(af1,    ldW(sW, 22 + nt, lane), acc);
        float b = bfo1[nt * 16 + m];
#pragma unroll
        for (int r = 0; r < 8; ++r)
          sH[w][(r + 8 * h) * 96 + nt * 16 + m] = f2bf(fmaxf(acc[r] + b, 0.0f));
      }
    }
    __syncthreads();

    // ---- fo layer2 + masked scatter (row < col) ----
    if (valid) {
      const bf16_t* fr = &sH[w][m * 96];
      v16bf f0 = fragA(fr, h), f1 = fragA(fr + 32, h);
#pragma unroll
      for (int nt = 0; nt < 2; ++nt) {
        v8f acc = {};
        acc = wmma_bf16(f0, ldW(sW, 26 + nt, lane), acc);
        acc = wmma_bf16(f1, ldW(sW, 28 + nt, lane), acc);
        float b = bfo2[nt * 16 + m];
#pragma unroll
        for (int r = 0; r < 8; ++r) {
          int i = r + 8 * h, gi = e0 + i;
          if (gi < E) {
            int er = sR[w][i], ec = sC[w][i];
            if (er < ec) atomicAdd(&flowOut[(size_t)er * 32 + nt * 16 + m], acc[r] + b);
          }
        }
      }
    }
    __syncthreads();

    // ---- fi layer1 (same A fragments) ----
    if (valid) {
#pragma unroll
      for (int nt = 0; nt < 4; ++nt) {
        v8f acc = {};
        acc = wmma_bf16(aflow0, ldW(sW, 30 + nt, lane), acc);
        acc = wmma_bf16(af1,    ldW(sW, 34 + nt, lane), acc);
        float b = bfi1[nt * 16 + m];
#pragma unroll
        for (int r = 0; r < 8; ++r)
          sH[w][(r + 8 * h) * 96 + nt * 16 + m] = f2bf(fmaxf(acc[r] + b, 0.0f));
      }
    }
    __syncthreads();

    // ---- fi layer2 + masked scatter (row > col) ----
    if (valid) {
      const bf16_t* fr = &sH[w][m * 96];
      v16bf f0 = fragA(fr, h), f1 = fragA(fr + 32, h);
#pragma unroll
      for (int nt = 0; nt < 2; ++nt) {
        v8f acc = {};
        acc = wmma_bf16(f0, ldW(sW, 38 + nt, lane), acc);
        acc = wmma_bf16(f1, ldW(sW, 40 + nt, lane), acc);
        float b = bfi2[nt * 16 + m];
#pragma unroll
        for (int r = 0; r < 8; ++r) {
          int i = r + 8 * h, gi = e0 + i;
          if (gi < E) {
            int er = sR[w][i], ec = sC[w][i];
            if (er > ec) atomicAdd(&flowIn[(size_t)er * 32 + nt * 16 + m], acc[r] + b);
          }
        }
      }
    }
    __syncthreads();
  }
}

// ---------------- K2: softmax denominator ----------------
__global__ void sum_kernel(const float* __restrict__ dec,
                           const int* __restrict__ row, const int* __restrict__ col,
                           const unsigned* __restrict__ maxOut, const unsigned* __restrict__ maxIn,
                           float* __restrict__ sumOut, float* __restrict__ sumIn, int E) {
  int e = blockIdx.x * blockDim.x + threadIdx.x;
  if (e >= E) return;
  int r = row[e], c = col[e];
  if (r == c) return;
  float d = dec[e];
  if (r < c) atomicAdd(&sumOut[r], __expf(d - fdec(maxOut[r])));
  else       atomicAdd(&sumIn[r],  __expf(d - fdec(maxIn[r])));
}

// ---------------- K3: attention-weighted scatter of x[col] ----------------
__global__ void attn_kernel(const float* __restrict__ x, const float* __restrict__ dec,
                            const int* __restrict__ row, const int* __restrict__ col,
                            const unsigned* __restrict__ maxOut, const unsigned* __restrict__ maxIn,
                            const float* __restrict__ sumOut, const float* __restrict__ sumIn,
                            float* __restrict__ attnOut, float* __restrict__ attnIn, int E) {
  int e = blockIdx.x * blockDim.x + threadIdx.x;
  if (e >= E) return;
  int r = row[e], c = col[e];
  if (r == c) return;
  float d = dec[e];
  float mx, s; float* acc;
  if (r < c) { mx = fdec(maxOut[r]); s = sumOut[r]; acc = attnOut + (size_t)r * 32; }
  else       { mx = fdec(maxIn[r]);  s = sumIn[r];  acc = attnIn  + (size_t)r * 32; }
  float wgt = __expf(d - mx) / fmaxf(s, 1e-16f);
  const float* xc = x + (size_t)c * 32;
#pragma unroll
  for (int f = 0; f < 32; ++f) atomicAdd(&acc[f], xc[f] * wgt);
}

// ---------------- K4: node linears (WMMA) ----------------
// Wn frags 0..3 (kc*2+nt), Wa frags 4..9 (kc*2+nt)
__global__ void __launch_bounds__(256)
node_kernel(const float* __restrict__ x,
            const float* __restrict__ flowOut, const float* __restrict__ flowIn,
            const float* __restrict__ attnOut, const float* __restrict__ attnIn,
            const float* __restrict__ Wn, const float* __restrict__ bn,
            const float* __restrict__ Wa, const float* __restrict__ ba,
            float* __restrict__ outNode, float* __restrict__ outAttn, int N) {
  __shared__ __align__(64) bf16_t sW[10 * 512];
  __shared__ __align__(64) bf16_t sB[8][16 * 96];
  packW(sW,            Wn, 64, 32, 2, 2);
  packW(sW + 4 * 512,  Wa, 96, 32, 3, 2);
  __syncthreads();

  const int tid = threadIdx.x, w = tid >> 5, lane = tid & 31, h = lane >> 4, m = lane & 15;
  const int T = (N + 15) / 16;

  for (int tb = blockIdx.x * 8; tb < T; tb += gridDim.x * 8) {
    const int tile = tb + w;
    const bool valid = tile < T;
    const int n0 = tile * 16;

    // ---- node_out = [flow_in | flow_out] @ Wn + bn ----
    if (valid) {
      int node = n0 + m; if (node >= N) node = N - 1;
      const float4* fi = (const float4*)(flowIn  + (size_t)node * 32 + 16 * h);
      const float4* fo = (const float4*)(flowOut + (size_t)node * 32 + 16 * h);
      bf16_t* br = &sB[w][m * 96];
      st8(br + 16 * h,          fi[0], fi[1]);
      st8(br + 16 * h + 8,      fi[2], fi[3]);
      st8(br + 32 + 16 * h,     fo[0], fo[1]);
      st8(br + 32 + 16 * h + 8, fo[2], fo[3]);
    }
    __syncthreads();
    if (valid) {
      const bf16_t* br = &sB[w][m * 96];
      v16bf c0 = fragA(br, h), c1 = fragA(br + 32, h);
#pragma unroll
      for (int nt = 0; nt < 2; ++nt) {
        v8f acc = {};
        acc = wmma_bf16(c0, ldW(sW, 0 + nt, lane), acc);
        acc = wmma_bf16(c1, ldW(sW, 2 + nt, lane), acc);
        float b = bn[nt * 16 + m];
#pragma unroll
        for (int r = 0; r < 8; ++r) {
          int gn = n0 + r + 8 * h;
          if (gn < N) outNode[(size_t)gn * 32 + nt * 16 + m] = acc[r] + b;
        }
      }
    }
    __syncthreads();

    // ---- attn_node = [x | attn_in | attn_out] @ Wa + ba ----
    if (valid) {
      int node = n0 + m; if (node >= N) node = N - 1;
      const float4* xp = (const float4*)(x       + (size_t)node * 32 + 16 * h);
      const float4* ai = (const float4*)(attnIn  + (size_t)node * 32 + 16 * h);
      const float4* ao = (const float4*)(attnOut + (size_t)node * 32 + 16 * h);
      bf16_t* br = &sB[w][m * 96];
      st8(br + 16 * h,          xp[0], xp[1]);
      st8(br + 16 * h + 8,      xp[2], xp[3]);
      st8(br + 32 + 16 * h,     ai[0], ai[1]);
      st8(br + 32 + 16 * h + 8, ai[2], ai[3]);
      st8(br + 64 + 16 * h,     ao[0], ao[1]);
      st8(br + 64 + 16 * h + 8, ao[2], ao[3]);
    }
    __syncthreads();
    if (valid) {
      const bf16_t* br = &sB[w][m * 96];
      v16bf c0 = fragA(br, h), c1 = fragA(br + 32, h), c2 = fragA(br + 64, h);
#pragma unroll
      for (int nt = 0; nt < 2; ++nt) {
        v8f acc = {};
        acc = wmma_bf16(c0, ldW(sW, 4 + nt, lane), acc);
        acc = wmma_bf16(c1, ldW(sW, 6 + nt, lane), acc);
        acc = wmma_bf16(c2, ldW(sW, 8 + nt, lane), acc);
        float b = ba[nt * 16 + m];
#pragma unroll
        for (int r = 0; r < 8; ++r) {
          int gn = n0 + r + 8 * h;
          if (gn < N) outAttn[(size_t)gn * 32 + nt * 16 + m] = acc[r] + b;
        }
      }
    }
    __syncthreads();
  }
}

// ---------------- host launcher ----------------
extern "C" void kernel_launch(void* const* d_in, const int* in_sizes, int n_in,
                              void* d_out, int out_size, void* d_ws, size_t ws_size,
                              hipStream_t stream) {
  const float* x     = (const float*)d_in[0];
  const float* eattr = (const float*)d_in[1];
  const int*   row   = (const int*)d_in[2];
  const int*   col   = (const int*)d_in[3];
  const float* We1 = (const float*)d_in[4];  const float* be1 = (const float*)d_in[5];
  const float* We2 = (const float*)d_in[6];  const float* be2 = (const float*)d_in[7];
  const float* Wfo1 = (const float*)d_in[8];  const float* bfo1 = (const float*)d_in[9];
  const float* Wfo2 = (const float*)d_in[10]; const float* bfo2 = (const float*)d_in[11];
  const float* Wfi1 = (const float*)d_in[12]; const float* bfi1 = (const float*)d_in[13];
  const float* Wfi2 = (const float*)d_in[14]; const float* bfi2 = (const float*)d_in[15];
  const float* Wn  = (const float*)d_in[16]; const float* bn  = (const float*)d_in[17];
  const float* Wc1 = (const float*)d_in[18]; const float* bc1 = (const float*)d_in[19];
  const float* Wc2 = (const float*)d_in[20]; const float* bc2 = (const float*)d_in[21];
  const float* Wa  = (const float*)d_in[22]; const float* ba  = (const float*)d_in[23];

  const int N = in_sizes[0] / 32;
  const int E = in_sizes[2];

  // d_out: [node_out N*32][e E*16][attn_node N*32][dec E]
  float* out      = (float*)d_out;
  float* outNode  = out;
  float* outE     = out + (size_t)N * 32;
  float* outAttn  = outE + (size_t)E * 16;
  float* outDec   = outAttn + (size_t)N * 32;

  // workspace: 4 accumulators N*32 f32, 2 max N u32, 2 sum N f32
  float*    ws      = (float*)d_ws;
  float*    flowOut = ws;
  float*    flowIn  = flowOut + (size_t)N * 32;
  float*    attnOut = flowIn  + (size_t)N * 32;
  float*    attnIn  = attnOut + (size_t)N * 32;
  unsigned* maxOut  = (unsigned*)(attnIn + (size_t)N * 32);
  unsigned* maxIn   = maxOut + N;
  float*    sumOut  = (float*)(maxIn + N);
  float*    sumIn   = sumOut + N;

  const size_t zTot = (size_t)N * 132;   // all accumulators + max(0-bits) + sums
  zero_kernel<<<2048, 256, 0, stream>>>(ws, zTot);

  edge_kernel<<<1250, 128, 0, stream>>>(
      x, eattr, row, col,
      We1, be1, We2, be2, Wfo1, bfo1, Wfo2, bfo2, Wfi1, bfi1, Wfi2, bfi2,
      Wc1, bc1, Wc2, bc2,
      outE, outDec, flowOut, flowIn, maxOut, maxIn, E);

  const int eb = (E + 255) / 256;
  sum_kernel<<<eb, 256, 0, stream>>>(outDec, row, col, maxOut, maxIn, sumOut, sumIn, E);
  attn_kernel<<<eb, 256, 0, stream>>>(x, outDec, row, col, maxOut, maxIn, sumOut, sumIn,
                                      attnOut, attnIn, E);

  const int Tn = (N + 15) / 16;
  node_kernel<<<(Tn + 7) / 8, 256, 0, stream>>>(
      x, flowOut, flowIn, attnOut, attnIn, Wn, bn, Wa, ba, outNode, outAttn, N);
}
